// TransformerDecoderLayer_70282844832283
// MI455X (gfx1250) — matrix operand order repaired and checked by reading, not compiled, not run on hardware
//
#include <hip/hip_runtime.h>
#include <math.h>

typedef __attribute__((ext_vector_type(16))) _Float16 v16h;
typedef __attribute__((ext_vector_type(8)))  float    v8f;

// ---------------------------------------------------------------------------
// Problem constants (match reference module constants)
// ---------------------------------------------------------------------------
#define DD   256
#define HH   8
#define HD_  32
#define LL   4
#define PP   4
#define BB   2
#define QQ   1024
#define NTOK (BB * QQ)          // 2048
#define VALID_PER_B 21760       // 128^2 + 64^2 + 32^2 + 16^2
#define NVALID (BB * VALID_PER_B)

// ---------------------------------------------------------------------------
// WMMA helpers (CDNA5 v_wmma_f32_16x16x32_f16)
// ---------------------------------------------------------------------------
__device__ __forceinline__ v8f wmma_f16(v16h a, v16h b, v8f c) {
  return __builtin_amdgcn_wmma_f32_16x16x32_f16(
      /*neg_a=*/false, a, /*neg_b=*/false, b,
      /*c_mod=*/(short)0, c, /*reuse_a=*/false, /*reuse_b=*/false);
}

// A fragment: 16x32 f16 tile, row-major with leading dim ld.
// lane&15 = row M; lane>>4 selects K 0..15 vs 16..31
__device__ __forceinline__ v16h frag_a16(const _Float16* s, int ld, int lane) {
  const _Float16* p = s + (lane & 15) * ld + ((lane >> 4) << 4);
  v16h a;
#pragma unroll
  for (int e = 0; e < 16; ++e) a[e] = p[e];
  return a;
}

// B fragment from "Bt" storage [N][K] (K contiguous), leading dim ld.
// lane&15 = N col; lane>>4 selects K 0..15 vs 16..31.
__device__ __forceinline__ v16h frag_bt16(const _Float16* s, int ld, int lane) {
  const _Float16* p = s + (lane & 15) * ld + ((lane >> 4) << 4);
  v16h b;
#pragma unroll
  for (int e = 0; e < 16; ++e) b[e] = p[e];
  return b;
}

// B fragment from row-major [K][N] storage (N contiguous), leading dim ld.
__device__ __forceinline__ v16h frag_brow(const _Float16* s, int ld, int lane) {
  const _Float16* p = s + ((lane >> 4) << 4) * ld + (lane & 15);
  v16h b;
#pragma unroll
  for (int e = 0; e < 16; ++e) b[e] = p[e * ld];
  return b;
}

__device__ __forceinline__ float gelu_f(float x) {
  const float k = 0.7978845608028654f;
  return 0.5f * x * (1.f + tanhf(k * (x + 0.044715f * x * x * x)));
}
__device__ __forceinline__ int imin(int a, int b) { return a < b ? a : b; }
__device__ __forceinline__ int imax(int a, int b) { return a > b ? a : b; }

// ---------------------------------------------------------------------------
// Generic WMMA GEMM: C[M,N] = act(A[M,K] @ W[K,N] + bias) (+ resid)
// block 256 thr = 8 waves; block tile 128x128; wave tile 32x64 (2x4 wmma)
// A rows optionally gathered via gidx (compacted feature-pyramid rows).
// Next-chunk global_prefetch_b8 issued to hide HBM latency behind WMMA.
// ---------------------------------------------------------------------------
template <bool GATHER, bool BIAS, bool RESID, bool GELU_, bool HALF_OUT>
__global__ void __launch_bounds__(256)
gemm_kernel(const float* __restrict__ A, const int* __restrict__ gidx,
            const float* __restrict__ W, const float* __restrict__ bias,
            const float* __restrict__ resid, void* __restrict__ Cout,
            int M, int N, int K) {
  __shared__ _Float16 As[128 * 32];
  __shared__ _Float16 Bs[128 * 32];  // transposed [n][k]
  const int tid = threadIdx.x;
  const int lane = tid & 31, wave = tid >> 5;
  const int bm = blockIdx.y * 128, bn = blockIdx.x * 128;
  const int wm = (wave >> 1) * 32, wn = (wave & 1) * 64;
  v8f acc[2][4] = {};

  for (int kc = 0; kc < K; kc += 32) {
    __syncthreads();
    for (int i = tid; i < 128 * 32; i += 256) {
      int r = i >> 5, c = i & 31;
      int gr = bm + r;
      float v = 0.f;
      if (gr < M) {
        long row = GATHER ? (long)gidx[gr] : (long)gr;
        v = A[row * (long)K + kc + c];
      }
      As[i] = (_Float16)v;
    }
    for (int i = tid; i < 128 * 32; i += 256) {
      int n = i >> 5, k = i & 31;
      Bs[i] = (_Float16)W[(long)(kc + k) * N + bn + n];
    }
    // Prefetch next K-chunk (A rows + B panel) into cache hierarchy.
    if (kc + 32 < K) {
      int gr = bm + (tid >> 1);
      if (gr < M) {
        long row = GATHER ? (long)gidx[gr] : (long)gr;
        __builtin_prefetch(&A[row * (long)K + kc + 32 + (tid & 1) * 16], 0, 1);
      }
      __builtin_prefetch(&W[(long)(kc + 32 + (tid >> 3)) * N + bn + (tid & 7) * 16], 0, 1);
    }
    __syncthreads();
    v16h a0 = frag_a16(As + (wm + 0) * 32, 32, lane);
    v16h a1 = frag_a16(As + (wm + 16) * 32, 32, lane);
#pragma unroll
    for (int ni = 0; ni < 4; ++ni) {
      v16h b = frag_bt16(Bs + (wn + 16 * ni) * 32, 32, lane);
      acc[0][ni] = wmma_f16(a0, b, acc[0][ni]);
      acc[1][ni] = wmma_f16(a1, b, acc[1][ni]);
    }
  }

  const int rsub = (lane >> 4) << 3;  // row offset within 16-tile
  const int csub = lane & 15;
#pragma unroll
  for (int mi = 0; mi < 2; ++mi) {
#pragma unroll
    for (int ni = 0; ni < 4; ++ni) {
#pragma unroll
      for (int j = 0; j < 8; ++j) {
        int row = bm + wm + 16 * mi + j + rsub;
        int col = bn + wn + 16 * ni + csub;
        if (row < M) {
          float v = acc[mi][ni][j];
          if (BIAS)  v += bias[col];
          if (GELU_) v = gelu_f(v);
          if (RESID) v += resid[(long)row * N + col];
          if (HALF_OUT) ((_Float16*)Cout)[(long)row * N + col] = (_Float16)v;
          else          ((float*)Cout)[(long)row * N + col] = v;
        }
      }
    }
  }
}

// ---------------------------------------------------------------------------
// LayerNorm over D=256, one block per row
// ---------------------------------------------------------------------------
__global__ void __launch_bounds__(256)
layernorm_kernel(const float* __restrict__ x, const float* __restrict__ w,
                 const float* __restrict__ bb, float* __restrict__ y) {
  __shared__ float red[256];
  int row = blockIdx.x, c = threadIdx.x;
  float v = x[(long)row * DD + c];
  red[c] = v;
  __syncthreads();
  for (int s = 128; s > 0; s >>= 1) { if (c < s) red[c] += red[c + s]; __syncthreads(); }
  float mean = red[0] / (float)DD;
  __syncthreads();
  float d = v - mean;
  red[c] = d * d;
  __syncthreads();
  for (int s = 128; s > 0; s >>= 1) { if (c < s) red[c] += red[c + s]; __syncthreads(); }
  float var = red[0] / (float)DD;
  y[(long)row * DD + c] = d * rsqrtf(var + 1e-5f) * w[c] + bb[c];
}

// ---------------------------------------------------------------------------
// Build compact-row -> padded-flat-row gather index for valid pyramid regions
// ---------------------------------------------------------------------------
__global__ void init_gather_kernel(int* __restrict__ gidx) {
  int i = blockIdx.x * 256 + threadIdx.x;
  if (i >= NVALID) return;
  int b = i / VALID_PER_B, r = i % VALID_PER_B;
  int l, hs;
  if (r < 16384)      { l = 0; hs = 128; }
  else if (r < 20480) { l = 1; hs = 64;  r -= 16384; }
  else if (r < 21504) { l = 2; hs = 32;  r -= 20480; }
  else                { l = 3; hs = 16;  r -= 21504; }
  int y = r / hs, x = r % hs;
  gidx[i] = ((b * LL + l) * 128 + y) * 128 + x;
}

// ---------------------------------------------------------------------------
// RoPE + pack q,k,v into [b*H+h][q][32] f16 (scale folded into q)
// ---------------------------------------------------------------------------
__global__ void __launch_bounds__(256)
sa_rope_kernel(const float* __restrict__ qkv, const float* __restrict__ qsp,
               _Float16* __restrict__ QH, _Float16* __restrict__ KH,
               _Float16* __restrict__ VH) {
  int t = blockIdx.x, idx = threadIdx.x;
  int h = idx >> 5, hd = idx & 31, pr = hd >> 1;
  int b = t >> 10, q = t & 1023;
  float py = qsp[t * 2 + 0] * 128.f, px = qsp[t * 2 + 1] * 128.f;
  float inv = powf(100.f, -(float)(pr < 8 ? pr : pr - 8) / 8.f);
  float ang = (pr < 8 ? py : px) * inv;
  float cs = cosf(ang), sn = sinf(ang);
  long base = (long)t * 768 + h * 32;
  float q0 = qkv[base + 2 * pr],       q1 = qkv[base + 2 * pr + 1];
  float k0 = qkv[base + 256 + 2 * pr], k1 = qkv[base + 256 + 2 * pr + 1];
  float vv = qkv[base + 512 + hd];
  float qo = (hd & 1) ? (q0 * sn + q1 * cs) : (q0 * cs - q1 * sn);
  float ko = (hd & 1) ? (k0 * sn + k1 * cs) : (k0 * cs - k1 * sn);
  const float scale = 0.17677669529663687f;  // 1/sqrt(32)
  long o = ((long)(b * HH + h) * QQ + q) * 32 + hd;
  QH[o] = (_Float16)(qo * scale);
  KH[o] = (_Float16)ko;
  VH[o] = (_Float16)vv;
}

// ---------------------------------------------------------------------------
// Flash-style self-attention: grid (B*H, Q/64), block 128 (4 waves, 16 q-rows each)
// ---------------------------------------------------------------------------
__global__ void __launch_bounds__(128)
sa_attn_kernel(const _Float16* __restrict__ QH, const _Float16* __restrict__ KH,
               const _Float16* __restrict__ VH, float* __restrict__ AO) {
  __shared__ _Float16 Ks[64 * 32];
  __shared__ _Float16 Vs[64 * 32];
  __shared__ _Float16 Ps[4][16 * 64];
  int bh = blockIdx.x, qb = blockIdx.y * 64;
  int b = bh >> 3, h = bh & 7;
  int tid = threadIdx.x, lane = tid & 31, wave = tid >> 5;
  const _Float16* Qp = QH + (long)bh * QQ * 32;
  const _Float16* Kp = KH + (long)bh * QQ * 32;
  const _Float16* Vp = VH + (long)bh * QQ * 32;

  v16h qa = frag_a16(Qp + (qb + wave * 16) * 32, 32, lane);
  float m[8], ls[8];
  v8f o0 = {}, o1 = {};
#pragma unroll
  for (int j = 0; j < 8; ++j) { m[j] = -1e30f; ls[j] = 0.f; }

  for (int kc = 0; kc < QQ; kc += 64) {
    __syncthreads();
    for (int i = tid; i < 64 * 32; i += 128) {
      Ks[i] = Kp[kc * 32 + i];
      Vs[i] = Vp[kc * 32 + i];
    }
    __syncthreads();

    v8f s0, s1, s2, s3;
    {
      v8f z = {};
      s0 = wmma_f16(qa, frag_bt16(Ks + 0 * 16 * 32, 32, lane), z);
      s1 = wmma_f16(qa, frag_bt16(Ks + 1 * 16 * 32, 32, lane), z);
      s2 = wmma_f16(qa, frag_bt16(Ks + 2 * 16 * 32, 32, lane), z);
      s3 = wmma_f16(qa, frag_bt16(Ks + 3 * 16 * 32, 32, lane), z);
    }
#pragma unroll
    for (int j = 0; j < 8; ++j) {
      float rmax = fmaxf(fmaxf(s0[j], s1[j]), fmaxf(s2[j], s3[j]));
#pragma unroll
      for (int d = 1; d < 16; d <<= 1) rmax = fmaxf(rmax, __shfl_xor(rmax, d, 32));
      float mn = fmaxf(m[j], rmax);
      float alpha = __expf(m[j] - mn);
      m[j] = mn;
      ls[j] *= alpha; o0[j] *= alpha; o1[j] *= alpha;
      int prow = (j + ((lane >> 4) << 3)) * 64 + (lane & 15);
      float p0 = __expf(s0[j] - mn), p1 = __expf(s1[j] - mn);
      float p2 = __expf(s2[j] - mn), p3 = __expf(s3[j] - mn);
      Ps[wave][prow + 0]  = (_Float16)p0;
      Ps[wave][prow + 16] = (_Float16)p1;
      Ps[wave][prow + 32] = (_Float16)p2;
      Ps[wave][prow + 48] = (_Float16)p3;
      float psum = p0 + p1 + p2 + p3;
#pragma unroll
      for (int d = 1; d < 16; d <<= 1) psum += __shfl_xor(psum, d, 32);
      ls[j] += psum;
    }
#pragma unroll
    for (int c = 0; c < 2; ++c) {
      v16h pa = frag_a16(&Ps[wave][c * 32], 64, lane);
      o0 = wmma_f16(pa, frag_brow(Vs + c * 32 * 32 + 0, 32, lane), o0);
      o1 = wmma_f16(pa, frag_brow(Vs + c * 32 * 32 + 16, 32, lane), o1);
    }
  }
#pragma unroll
  for (int j = 0; j < 8; ++j) {
    int row = qb + wave * 16 + j + ((lane >> 4) << 3);
    long ob = ((long)(b * QQ + row)) * DD + h * 32;
    float inv = 1.f / ls[j];
    AO[ob + (lane & 15)]      = o0[j] * inv;
    AO[ob + 16 + (lane & 15)] = o1[j] * inv;
  }
}

// ---------------------------------------------------------------------------
// Softmax over 16 deformable weights per (token, head)
// ---------------------------------------------------------------------------
__global__ void softmax16_kernel(float* __restrict__ a) {
  int i = blockIdx.x * 256 + threadIdx.x;  // token*8 + h
  if (i >= NTOK * HH) return;
  float* p = a + (long)i * 16;
  float mx = -1e30f;
#pragma unroll
  for (int k = 0; k < 16; ++k) mx = fmaxf(mx, p[k]);
  float e[16], s = 0.f;
#pragma unroll
  for (int k = 0; k < 16; ++k) { e[k] = __expf(p[k] - mx); s += e[k]; }
  float inv = 1.f / s;
#pragma unroll
  for (int k = 0; k < 16; ++k) p[k] = e[k] * inv;
}

// ---------------------------------------------------------------------------
// Deformable sampling: block per token, thread = h*32+hd
// ---------------------------------------------------------------------------
__global__ void __launch_bounds__(256)
md_sample_kernel(const float* __restrict__ OFF, const float* __restrict__ AW,
                 const float* __restrict__ qsp, const _Float16* __restrict__ VP,
                 float* __restrict__ out) {
  const int lvoff[4] = {0, 16384, 20480, 21504};
  const int lhs[4]   = {128, 64, 32, 16};
  int t = blockIdx.x, idx = threadIdx.x;
  int h = idx >> 5, hd = idx & 31;
  int b = t >> 10;
  float py = qsp[t * 2], px = qsp[t * 2 + 1];
  float acc = 0.f;
  int ch = h * 32 + hd;
  for (int l = 0; l < 4; ++l) {
    int hs = lhs[l];
    float hf = (float)hs;
    long vbase = ((long)b * VALID_PER_B + lvoff[l]) * DD;
    for (int p = 0; p < 4; ++p) {
      long ob = (long)t * 256 + ((h * 4 + l) * 4 + p) * 2;
      float ly = py + OFF[ob] / hf;
      float lx = px + OFF[ob + 1] / hf;
      float aw = AW[(long)t * 128 + h * 16 + l * 4 + p];
      float yy = ly * hf - 0.5f, xx = lx * hf - 0.5f;
      float y0 = floorf(yy), x0 = floorf(xx);
      float wy = yy - y0, wx = xx - x0;
      int y0i = imin(imax((int)y0, 0), hs - 1);
      int y1i = imin(imax((int)y0 + 1, 0), hs - 1);
      int x0i = imin(imax((int)x0, 0), hs - 1);
      int x1i = imin(imax((int)x0 + 1, 0), hs - 1);
      float f00 = (float)VP[vbase + (long)(y0i * hs + x0i) * DD + ch];
      float f01 = (float)VP[vbase + (long)(y0i * hs + x1i) * DD + ch];
      float f10 = (float)VP[vbase + (long)(y1i * hs + x0i) * DD + ch];
      float f11 = (float)VP[vbase + (long)(y1i * hs + x1i) * DD + ch];
      float sv = f00 * (1.f - wy) * (1.f - wx) + f01 * (1.f - wy) * wx +
                 f10 * wy * (1.f - wx) + f11 * wy * wx;
      acc += sv * aw;
    }
  }
  out[(long)t * DD + idx] = acc;
}

// ---------------------------------------------------------------------------
// Neighborhood attention: one wave per (token, head); lane = hd channel
// ---------------------------------------------------------------------------
__global__ void __launch_bounds__(128)
na_attn_kernel(const float* __restrict__ QN, const float* __restrict__ qsp,
               const _Float16* __restrict__ KVP, float* __restrict__ out) {
  const int lvoff[4] = {0, 16384, 20480, 21504};
  const int lhs[4]   = {128, 64, 32, 16};
  const int lr[4]    = {1, 2, 3, 4};
  int g = blockIdx.x * 4 + (threadIdx.x >> 5);  // (token,head) id
  int lane = threadIdx.x & 31;
  int t = g >> 3, h = g & 7;
  int b = t >> 10;
  float py = qsp[t * 2], px = qsp[t * 2 + 1];
  const float scale = 0.17677669529663687f;
  float qb = QN[(long)t * DD + h * 32 + lane] * scale;
  int pr = lane >> 1;
  bool yhalf = pr < 8;
  float invf = powf(100.f, -(float)(yhalf ? pr : pr - 8) / 8.f);
  float m = -1e30f, ssum = 0.f, acc = 0.f;

  for (int l = 0; l < 4; ++l) {
    int hs = lhs[l];
    float hf = (float)hs;
    // RoPE q for this level
    float aq = (yhalf ? py * hf : px * hf) * invf;
    float cq = cosf(aq), sq = sinf(aq);
    float qpart = __shfl_xor(qb, 1, 32);
    float ql = (lane & 1) ? (qpart * sq + qb * cq) : (qb * cq - qpart * sq);
    int cy = (int)floorf(py * hf), cx = (int)floorf(px * hf);
    int r = lr[l];
    long base = (long)b * VALID_PER_B + lvoff[l];
    for (int dy = -r; dy <= r; ++dy) {
      for (int dx = -r; dx <= r; ++dx) {
        int ky = cy + dy, kx = cx + dx;
        bool valid = (ky >= 0) && (ky < hs) && (kx >= 0) && (kx < hs);
        int yi = imin(imax(ky, 0), hs - 1);
        int xi = imin(imax(kx, 0), hs - 1);
        long rowb = (base + yi * hs + xi) * 512;
        float kf = (float)KVP[rowb + h * 32 + lane];
        float vf = (float)KVP[rowb + 256 + h * 32 + lane];
        float ak = (yhalf ? ((float)ky + 0.5f) : ((float)kx + 0.5f)) * invf;
        float ck = cosf(ak), sk = sinf(ak);
        float kpart = __shfl_xor(kf, 1, 32);
        float kl = (lane & 1) ? (kpart * sk + kf * ck) : (kf * ck - kpart * sk);
        float lg = ql * kl;
#pragma unroll
        for (int d = 1; d < 32; d <<= 1) lg += __shfl_xor(lg, d, 32);
        lg = valid ? lg : -1e9f;
        float mn = fmaxf(m, lg);
        float al = __expf(m - mn);
        float p = __expf(lg - mn);
        m = mn;
        ssum = ssum * al + p;
        acc = acc * al + p * vf;
      }
    }
  }
  out[(long)t * DD + h * 32 + lane] = acc / ssum;
}

// ---------------------------------------------------------------------------
// Host launcher
// ---------------------------------------------------------------------------
extern "C" void kernel_launch(void* const* d_in, const int* in_sizes, int n_in,
                              void* d_out, int out_size, void* d_ws, size_t ws_size,
                              hipStream_t stream) {
  (void)in_sizes; (void)n_in; (void)out_size; (void)ws_size;
  const float* queries = (const float*)d_in[0];
  const float* qsp     = (const float*)d_in[2];
  const float* sfm     = (const float*)d_in[3];
  const float* sa_nw   = (const float*)d_in[5];
  const float* sa_nb   = (const float*)d_in[6];
  const float* sa_wqkv = (const float*)d_in[7];
  const float* sa_wo   = (const float*)d_in[8];
  const float* md_nw   = (const float*)d_in[9];
  const float* md_nb   = (const float*)d_in[10];
  const float* md_wv   = (const float*)d_in[11];
  const float* md_woff = (const float*)d_in[12];
  const float* md_boff = (const float*)d_in[13];
  const float* md_wattn= (const float*)d_in[14];
  const float* md_battn= (const float*)d_in[15];
  const float* md_wo   = (const float*)d_in[16];
  const float* na_nw   = (const float*)d_in[17];
  const float* na_nb   = (const float*)d_in[18];
  const float* na_wq   = (const float*)d_in[19];
  const float* na_wkv  = (const float*)d_in[20];
  const float* na_wo   = (const float*)d_in[21];
  const float* ffn_nw  = (const float*)d_in[22];
  const float* ffn_nb  = (const float*)d_in[23];
  const float* ffn_w1  = (const float*)d_in[24];
  const float* ffn_b1  = (const float*)d_in[25];
  const float* ffn_w2  = (const float*)d_in[26];
  const float* ffn_b2  = (const float*)d_in[27];

  size_t off = 0;
  auto take = [&](size_t bytes) -> void* {
    void* p = (char*)d_ws + off;
    off += (bytes + 255) & ~(size_t)255;
    return p;
  };
  float*    X    = (float*)take((size_t)NTOK * DD * 4);
  float*    Y    = (float*)take((size_t)NTOK * DD * 4);
  float*    QKV  = (float*)take((size_t)NTOK * 768 * 4);
  _Float16* QH   = (_Float16*)take((size_t)BB * HH * QQ * 32 * 2);
  _Float16* KH   = (_Float16*)take((size_t)BB * HH * QQ * 32 * 2);
  _Float16* VH   = (_Float16*)take((size_t)BB * HH * QQ * 32 * 2);
  float*    AO   = (float*)take((size_t)NTOK * DD * 4);
  float*    OFFb = (float*)take((size_t)NTOK * 256 * 4);
  float*    AWL  = (float*)take((size_t)NTOK * 128 * 4);
  int*      GIDX = (int*)take((size_t)NVALID * 4);
  _Float16* VP   = (_Float16*)take((size_t)NVALID * 256 * 2);
  _Float16* KVP  = (_Float16*)take((size_t)NVALID * 512 * 2);
  float*    HID  = (float*)take((size_t)NTOK * 1024 * 4);
  float*    QN   = QKV;  // reuse (free after rope pack)

  dim3 blk256(256), blk128(128);
  dim3 g_tok(NTOK);
  auto ggrid = [](int M, int N) { return dim3(N / 128, (M + 127) / 128); };

  // gather index for valid pyramid rows
  init_gather_kernel<<<(NVALID + 255) / 256, blk256, 0, stream>>>(GIDX);

  // ---- self attention ----
  layernorm_kernel<<<g_tok, blk256, 0, stream>>>(queries, sa_nw, sa_nb, Y);
  gemm_kernel<false,false,false,false,false><<<ggrid(NTOK,768), blk256, 0, stream>>>(
      Y, nullptr, sa_wqkv, nullptr, nullptr, QKV, NTOK, 768, 256);
  sa_rope_kernel<<<g_tok, blk256, 0, stream>>>(QKV, qsp, QH, KH, VH);
  sa_attn_kernel<<<dim3(BB * HH, QQ / 64), blk128, 0, stream>>>(QH, KH, VH, AO);
  gemm_kernel<false,false,true,false,false><<<ggrid(NTOK,256), blk256, 0, stream>>>(
      AO, nullptr, sa_wo, nullptr, queries, X, NTOK, 256, 256);

  // ---- deformable attention ----
  layernorm_kernel<<<g_tok, blk256, 0, stream>>>(X, md_nw, md_nb, Y);
  gemm_kernel<false,true,false,false,false><<<ggrid(NTOK,256), blk256, 0, stream>>>(
      Y, nullptr, md_woff, md_boff, nullptr, OFFb, NTOK, 256, 256);
  gemm_kernel<false,true,false,false,false><<<ggrid(NTOK,128), blk256, 0, stream>>>(
      Y, nullptr, md_wattn, md_battn, nullptr, AWL, NTOK, 128, 256);
  softmax16_kernel<<<(NTOK * HH + 255) / 256, blk256, 0, stream>>>(AWL);
  gemm_kernel<true,false,false,false,true><<<ggrid(NVALID,256), blk256, 0, stream>>>(
      sfm, GIDX, md_wv, nullptr, nullptr, VP, NVALID, 256, 256);
  md_sample_kernel<<<g_tok, blk256, 0, stream>>>(OFFb, AWL, qsp, VP, AO);
  gemm_kernel<false,false,true,false,false><<<ggrid(NTOK,256), blk256, 0, stream>>>(
      AO, nullptr, md_wo, nullptr, X, X, NTOK, 256, 256);

  // ---- neighborhood attention ----
  layernorm_kernel<<<g_tok, blk256, 0, stream>>>(X, na_nw, na_nb, Y);
  gemm_kernel<false,false,false,false,false><<<ggrid(NTOK,256), blk256, 0, stream>>>(
      Y, nullptr, na_wq, nullptr, nullptr, QN, NTOK, 256, 256);
  gemm_kernel<true,false,false,false,true><<<ggrid(NVALID,512), blk256, 0, stream>>>(
      sfm, GIDX, na_wkv, nullptr, nullptr, KVP, NVALID, 512, 256);
  na_attn_kernel<<<dim3(NTOK * HH / 4), blk128, 0, stream>>>(QN, qsp, KVP, AO);
  gemm_kernel<false,false,true,false,false><<<ggrid(NTOK,256), blk256, 0, stream>>>(
      AO, nullptr, na_wo, nullptr, X, X, NTOK, 256, 256);

  // ---- FFN ----
  layernorm_kernel<<<g_tok, blk256, 0, stream>>>(X, ffn_nw, ffn_nb, Y);
  gemm_kernel<false,true,false,true,false><<<ggrid(NTOK,1024), blk256, 0, stream>>>(
      Y, nullptr, ffn_w1, ffn_b1, nullptr, HID, NTOK, 1024, 256);
  gemm_kernel<false,true,true,false,false><<<ggrid(NTOK,256), blk256, 0, stream>>>(
      HID, nullptr, ffn_w2, ffn_b2, X, (float*)d_out, NTOK, 256, 1024);
}
